// TaskAlignedAssigner_12317966205418
// MI455X (gfx1250) — compile-verified, matrix operand order checked by reading, not run on hardware
//
#include <hip/hip_runtime.h>
#include <hip/hip_bf16.h>

// ---------------- problem constants ----------------
#define BS   16
#define NA   8400
#define NM   64
#define NC   80
#define TOPK 10
#define BG   80
#define EPS  1e-9f
#define CEPS 1e-7f
#define INV_PI2_4 0.4052847345693511f   // 4 / pi^2

typedef __attribute__((ext_vector_type(2)))  float    v2f;
typedef __attribute__((ext_vector_type(8)))  float    v8f;
typedef __attribute__((ext_vector_type(16))) _Float16 v16h;

#if defined(__has_builtin)
#  if __has_builtin(__builtin_amdgcn_wmma_f32_16x16x4_f32)
#    define HAVE_WMMA_F32X4 1
#  endif
#endif

// =====================================================================
// Kernel 1: one wave per 16x16 (gt x anchor) tile.
//   bbox_scores tile = onehot(labels) @ pd_scores^T  via WMMA,
//   then per-lane CIoU epilogue on the accumulator layout
//   (lane = anchor col, acc[v] = gt row).  Writes align & overlaps.
// =====================================================================
__global__ __launch_bounds__(32)
void k1_align(const float* __restrict__ pd_scores,
              const float* __restrict__ pd_bboxes,
              const float* __restrict__ anc,
              const int*   __restrict__ gt_labels,
              const float* __restrict__ gt_bboxes,
              const unsigned char* __restrict__ mask_gt,
              float* __restrict__ w_align,
              float* __restrict__ w_ov) {
    const int lane = threadIdx.x;        // 0..31
    const int a0   = blockIdx.x * 16;
    const int m0   = blockIdx.y * 16;
    const int b    = blockIdx.z;
    const int n    = lane & 15;          // anchor column within tile
    const int hi   = lane >> 4;          // lane half (K-split per ISA layout)
    const int a    = a0 + n;

    // label for GT row (m0 + n); lanes l and l+16 carry the same M rows.
    int lbl = gt_labels[(size_t)b * NM + m0 + n];
    lbl = lbl < 0 ? 0 : (lbl > NC - 1 ? NC - 1 : lbl);

    const float* srow = pd_scores + ((size_t)b * NA + a) * NC;
    v8f acc = {};

#ifdef HAVE_WMMA_F32X4
    // A: 16x4 one-hot fp32 (lane M=l%16, VGPR v -> K = 2*hi + v)
    // B: 4x16 fp32 (lane N=l%16, VGPR v -> K = 2*hi + v): float2 load.
    #pragma unroll
    for (int c0 = 0; c0 < NC; c0 += 4) {
        const int kb = c0 + 2 * hi;
        v2f Am, Bm;
        Am.x = (lbl == kb)     ? 1.0f : 0.0f;
        Am.y = (lbl == kb + 1) ? 1.0f : 0.0f;
        Bm.x = srow[kb];
        Bm.y = srow[kb + 1];
        acc = __builtin_amdgcn_wmma_f32_16x16x4_f32(
            false, Am, false, Bm, (short)0, acc, false, false);
    }
#else
    // Confirmed path: V_WMMA_F32_16X16X32_F16, one-hot exact in f16.
    // 16-bit A layout: half h (vgpr v=h/2, pos p=h&1):
    //   K = c0 + (v>=4 ? 16 : 0) + 2*(v&3) + p + hi*8
    for (int c0 = 0; c0 < 96; c0 += 32) {
        v16h Am, Bm;
        #pragma unroll
        for (int h = 0; h < 16; ++h) {
            const int v = h >> 1, p = h & 1;
            const int k = c0 + ((v >= 4) ? 16 : 0) + 2 * (v & 3) + p + hi * 8;
            Am[h] = (_Float16)((lbl == k) ? 1.0f : 0.0f);
            Bm[h] = (k < NC) ? (_Float16)srow[k] : (_Float16)0.0f;
        }
        acc = __builtin_amdgcn_wmma_f32_16x16x32_f16(
            false, Am, false, Bm, (short)0, acc, false, false);
    }
#endif

    // ---- CIoU epilogue: this lane owns anchor `a`, GT rows hi*8+v ----
    const float* pb = pd_bboxes + ((size_t)b * NA + a) * 4;
    const float px1 = pb[0], py1 = pb[1], px2 = pb[2], py2 = pb[3];
    const float ax  = anc[(size_t)a * 2 + 0];
    const float ay  = anc[(size_t)a * 2 + 1];
    const float w2  = px2 - px1, h2 = py2 - py1;
    const float at2 = atanf(w2 / (h2 + CEPS));

    #pragma unroll
    for (int v = 0; v < 8; ++v) {
        const int m = m0 + hi * 8 + v;
        const float* gb = gt_bboxes + ((size_t)b * NM + m) * 4;
        const float gx1 = gb[0], gy1 = gb[1], gx2 = gb[2], gy2 = gb[3];

        const bool ing = (ax - gx1 > EPS) && (ay - gy1 > EPS) &&
                         (gx2 - ax > EPS) && (gy2 - ay > EPS);
        const bool valid = ing && (mask_gt[(size_t)b * NM + m] != 0);

        const float w1 = gx2 - gx1, h1 = gy2 - gy1;
        const float iw = fminf(gx2, px2) - fmaxf(gx1, px1);
        const float ih = fminf(gy2, py2) - fmaxf(gy1, py1);
        const float inter = fmaxf(iw, 0.0f) * fmaxf(ih, 0.0f);
        const float uni   = w1 * h1 + w2 * h2 - inter + CEPS;
        const float iou   = inter / uni;
        const float cw = fmaxf(gx2, px2) - fminf(gx1, px1);
        const float ch = fmaxf(gy2, py2) - fminf(gy1, py1);
        const float c2 = cw * cw + ch * ch + CEPS;
        const float dx = (px1 + px2 - gx1 - gx2);
        const float dy = (py1 + py2 - gy1 - gy2);
        const float rho2 = (dx * dx + dy * dy) * 0.25f;
        const float da = atanf(w1 / (h1 + CEPS)) - at2;
        const float vv = INV_PI2_4 * da * da;
        const float alpha = vv / (vv - iou + 1.0f + CEPS);
        const float ciou  = iou - (rho2 / c2 + vv * alpha);

        const float ov  = valid ? fmaxf(ciou, 0.0f) : 0.0f;
        const float sc  = fmaxf(acc[v], 0.0f);
        const float ov2 = ov * ov;
        const float ov6 = ov2 * ov2 * ov2;
        const float al  = valid ? sqrtf(sc) * ov6 : 0.0f;

        const size_t off = ((size_t)b * NM + m) * (size_t)NA + a;
        w_align[off] = al;
        w_ov[off]    = ov;
    }
}

// =====================================================================
// Kernel 2: per (b,m) row: 10 sequential argmax passes -> mask_pos v1.
// =====================================================================
__global__ __launch_bounds__(256)
void k2_topk(const float* __restrict__ align,
             unsigned char* __restrict__ mask,
             const unsigned char* __restrict__ mask_gt) {
    const int r = blockIdx.x;                 // b*NM + m
    const float* row = align + (size_t)r * NA;
    unsigned char* mrow = mask + (size_t)r * NA;
    __shared__ float s_val[256];
    __shared__ int   s_idx[256];
    __shared__ int   chosen[TOPK];
    const int t = threadIdx.x;

    for (int a = t; a < NA; a += 256) mrow[a] = 0;
    __syncthreads();
    if (mask_gt[r] == 0) return;              // uniform per block

    for (int it = 0; it < TOPK; ++it) {
        float bv = -1.0f; int bi = 0x7fffffff;
        for (int a = t; a < NA; a += 256) {
            bool excl = false;
            for (int j = 0; j < it; ++j) excl = excl || (chosen[j] == a);
            if (excl) continue;
            const float v = row[a];
            if (v > bv || (v == bv && a < bi)) { bv = v; bi = a; }
        }
        s_val[t] = bv; s_idx[t] = bi;
        __syncthreads();
        for (int s = 128; s > 0; s >>= 1) {
            if (t < s) {
                const float ov = s_val[t + s]; const int oi = s_idx[t + s];
                if (ov > s_val[t] || (ov == s_val[t] && oi < s_idx[t])) {
                    s_val[t] = ov; s_idx[t] = oi;
                }
            }
            __syncthreads();
        }
        if (t == 0) {
            chosen[it] = s_idx[0];
            if (s_val[0] > 0.0f && s_idx[0] < NA) mrow[s_idx[0]] = 1;
        }
        __syncthreads();
    }
}

// =====================================================================
// Kernel 3: per anchor: fg count, multi-assignment resolution by
// overlaps argmax, gathers of labels / boxes / gt index.
// =====================================================================
__global__ __launch_bounds__(256)
void k3_resolve(const float* __restrict__ ov,
                unsigned char* __restrict__ mask,
                const int*   __restrict__ gt_labels,
                const float* __restrict__ gt_bboxes,
                float* __restrict__ o_tl, float* __restrict__ o_tb,
                float* __restrict__ o_fg, float* __restrict__ o_gi) {
    const int a = blockIdx.x * 256 + threadIdx.x;
    const int b = blockIdx.y;
    if (a >= NA) return;
    const size_t base = (size_t)b * NM;

    int fg = 0, first_m = 0, best_m = 0;
    float best_ov = -1.0f;
    for (int m = 0; m < NM; ++m) {
        const size_t off = (base + m) * (size_t)NA + a;
        if (mask[off]) { if (fg == 0) first_m = m; ++fg; }
        const float o = ov[off];
        if (o > best_ov) { best_ov = o; best_m = m; }   // first max on ties
    }
    int tgt; bool fgb;
    if (fg > 1) {
        for (int m = 0; m < NM; ++m)
            mask[(base + m) * (size_t)NA + a] = (m == best_m) ? 1 : 0;
        tgt = best_m; fgb = true;
    } else {
        tgt = (fg == 1) ? first_m : 0; fgb = (fg == 1);
    }
    const int lab = gt_labels[base + tgt];
    const size_t oa = (size_t)b * NA + a;
    o_tl[oa] = fgb ? (float)lab : (float)BG;
    o_gi[oa] = (float)tgt;
    o_fg[oa] = fgb ? 1.0f : 0.0f;
    const float* gb = gt_bboxes + (base + tgt) * 4;
    float* tb = o_tb + oa * 4;
    tb[0] = gb[0]; tb[1] = gb[1]; tb[2] = gb[2]; tb[3] = gb[3];
}

// =====================================================================
// Kernel 4: per (b,m) row: pos_align / pos_overlap (masked row maxima).
// =====================================================================
__global__ __launch_bounds__(256)
void k4_rowmax(const float* __restrict__ align, const float* __restrict__ ov,
               const unsigned char* __restrict__ mask,
               float* __restrict__ posal, float* __restrict__ posov) {
    const int r = blockIdx.x;
    __shared__ float sa[256], so[256];
    const int t = threadIdx.x;
    float ba = 0.0f, bo = 0.0f;
    for (int a = t; a < NA; a += 256) {
        const size_t off = (size_t)r * NA + a;
        if (mask[off]) { ba = fmaxf(ba, align[off]); bo = fmaxf(bo, ov[off]); }
    }
    sa[t] = ba; so[t] = bo;
    __syncthreads();
    for (int s = 128; s > 0; s >>= 1) {
        if (t < s) { sa[t] = fmaxf(sa[t], sa[t + s]); so[t] = fmaxf(so[t], so[t + s]); }
        __syncthreads();
    }
    if (t == 0) { posal[r] = sa[0]; posov[r] = so[0]; }
}

// =====================================================================
// Kernel 5a: per anchor norm = max_m mask*align*pos_ov/(pos_align+EPS)
// =====================================================================
__global__ __launch_bounds__(256)
void k5_norm(const float* __restrict__ align, const unsigned char* __restrict__ mask,
             const float* __restrict__ posal, const float* __restrict__ posov,
             float* __restrict__ norm) {
    const int a = blockIdx.x * 256 + threadIdx.x;
    const int b = blockIdx.y;
    if (a >= NA) return;
    float nmax = 0.0f;
    for (int m = 0; m < NM; ++m) {
        const int r = b * NM + m;
        const size_t off = (size_t)r * NA + a;
        if (mask[off])
            nmax = fmaxf(nmax, align[off] * posov[r] / (posal[r] + EPS));
    }
    norm[(size_t)b * NA + a] = nmax;
}

// =====================================================================
// Kernel 5b: coalesced one-hot fill of target_scores (B,A,C).
// =====================================================================
__global__ __launch_bounds__(256)
void k5_scores(const float* __restrict__ norm, const float* __restrict__ o_tl,
               const float* __restrict__ o_fg, float* __restrict__ o_ts) {
    const size_t i = (size_t)blockIdx.x * 256 + threadIdx.x;
    if (i >= (size_t)BS * NA * NC) return;
    const int    c  = (int)(i % NC);
    const size_t ba = i / NC;
    float v = 0.0f;
    if (o_fg[ba] > 0.0f && (int)o_tl[ba] == c) v = norm[ba];
    o_ts[i] = v;
}

// =====================================================================
extern "C" void kernel_launch(void* const* d_in, const int* in_sizes, int n_in,
                              void* d_out, int out_size, void* d_ws, size_t ws_size,
                              hipStream_t stream) {
    const float*         pd_scores = (const float*)d_in[0];   // (BS,NA,NC)
    const float*         pd_bboxes = (const float*)d_in[1];   // (BS,NA,4)
    const float*         anc       = (const float*)d_in[2];   // (NA,2)
    const int*           gt_labels = (const int*)d_in[3];     // (BS,NM,1)
    const float*         gt_bboxes = (const float*)d_in[4];   // (BS,NM,4)
    const unsigned char* mask_gt   = (const unsigned char*)d_in[5]; // (BS,NM,1)

    // workspace carve-up (~78 MB; entire working set fits in 192 MB L2)
    const size_t SZF = (size_t)BS * NM * NA;                  // 8,601,600
    float*         w_align = (float*)d_ws;
    float*         w_ov    = w_align + SZF;
    unsigned char* w_mask  = (unsigned char*)(w_ov + SZF);
    float*         w_posal = (float*)(w_mask + SZF);
    float*         w_posov = w_posal + (size_t)BS * NM;
    float*         w_norm  = w_posov + (size_t)BS * NM;       // BS*NA floats

    // output carve-up (flat, return order)
    float* out  = (float*)d_out;
    float* o_tl = out;                          // (BS,NA)
    float* o_tb = o_tl + (size_t)BS * NA;       // (BS,NA,4)
    float* o_ts = o_tb + (size_t)BS * NA * 4;   // (BS,NA,NC)
    float* o_fg = o_ts + (size_t)BS * NA * NC;  // (BS,NA)
    float* o_gi = o_fg + (size_t)BS * NA;       // (BS,NA)

    // K1: WMMA one-hot gather + CIoU epilogue, one wave per 16x16 tile
    k1_align<<<dim3(NA / 16, NM / 16, BS), dim3(32), 0, stream>>>(
        pd_scores, pd_bboxes, anc, gt_labels, gt_bboxes, mask_gt, w_align, w_ov);

    // K2: per-GT-row top-10
    k2_topk<<<dim3(BS * NM), dim3(256), 0, stream>>>(w_align, w_mask, mask_gt);

    // K3: per-anchor resolution + gathers
    k3_resolve<<<dim3((NA + 255) / 256, BS), dim3(256), 0, stream>>>(
        w_ov, w_mask, gt_labels, gt_bboxes, o_tl, o_tb, o_fg, o_gi);

    // K4: masked row maxima
    k4_rowmax<<<dim3(BS * NM), dim3(256), 0, stream>>>(
        w_align, w_ov, w_mask, w_posal, w_posov);

    // K5a: per-anchor norm
    k5_norm<<<dim3((NA + 255) / 256, BS), dim3(256), 0, stream>>>(
        w_align, w_mask, w_posal, w_posov, w_norm);

    // K5b: coalesced target_scores fill
    const size_t tot = (size_t)BS * NA * NC;
    k5_scores<<<dim3((unsigned)((tot + 255) / 256)), dim3(256), 0, stream>>>(
        w_norm, o_tl, o_fg, o_ts);
}
